// GoldFinchGPTAlphaGoCo_83958020702208
// MI455X (gfx1250) — matrix operand order repaired and checked by reading, not compiled, not run on hardware
//
#include <hip/hip_runtime.h>
#include <math.h>
#include <stdint.h>

// ---------------------------------------------------------------------------
// Problem constants (match reference)
// ---------------------------------------------------------------------------
#define Bc 2
#define Sc 2048
#define Hc 2048
#define NHc 32
#define KDc 64      // Hc / NHc
#define DMIX 32
#define DVAL 128

typedef __attribute__((ext_vector_type(16))) __bf16 v16bf;
typedef __attribute__((ext_vector_type(8)))  float  v8f;

union FragAB {
  v16bf v;
  float4 f4[2];
  unsigned short u[16];
};

static __device__ __forceinline__ unsigned short f32_to_bf16(float f) {
  union { float f; unsigned u; } x; x.f = f;
  unsigned r = x.u + 0x7fffu + ((x.u >> 16) & 1u);   // round-to-nearest-even
  return (unsigned short)(r >> 16);
}

// block of 256 threads (8 wave32) sum-reduce
static __device__ __forceinline__ float block_sum(float v, float* red) {
#pragma unroll
  for (int off = 16; off > 0; off >>= 1) v += __shfl_xor(v, off, 32);
  const int lane = threadIdx.x & 31, wv = threadIdx.x >> 5;
  __syncthreads();
  if (lane == 0) red[wv] = v;
  __syncthreads();
  if (wv == 0) {
    float r = (lane < 8) ? red[lane] : 0.f;
#pragma unroll
    for (int off = 4; off > 0; off >>= 1) r += __shfl_xor(r, off, 32);
    if (lane == 0) red[0] = r;
  }
  __syncthreads();
  return red[0];
}

// ---------------------------------------------------------------------------
// K1: xo_n = rms_norm(xo); shift_state = x[:, -1]
// ---------------------------------------------------------------------------
__global__ void rms_shift_kernel(const float* __restrict__ x,
                                 const float* __restrict__ xo,
                                 float* __restrict__ xo_n,
                                 float* __restrict__ shift_out) {
  __shared__ float red[8];
  const size_t token = blockIdx.x;
  const float* row = xo + token * (size_t)Hc;
  float ss = 0.f;
  for (int i = threadIdx.x; i < Hc; i += 256) { float t = row[i]; ss += t * t; }
  float inv = rsqrtf(block_sum(ss, red) / (float)Hc + 1e-6f);
  float* outr = xo_n + token * (size_t)Hc;
  for (int i = threadIdx.x; i < Hc; i += 256) outr[i] = row[i] * inv;
  const int s = (int)(token % Sc), b = (int)(token / Sc);
  if (s == Sc - 1) {
    const float* xr = x + token * (size_t)Hc;
    for (int i = threadIdx.x; i < Hc; i += 256) shift_out[(size_t)b * Hc + i] = xr[i];
  }
}

// ---------------------------------------------------------------------------
// K2: xxx = x + dxprev*maa_x ; xx2 = xo_n + dxo_prev*maa_vc ;
//     mq_pre = tanh(xxx@q_w1) ; t64 = tanh(xx2@kv_w1)
// ---------------------------------------------------------------------------
__global__ void mlp1_kernel(const float* __restrict__ x,
                            const float* __restrict__ shift_in,
                            const float* __restrict__ xo_n,
                            const float* __restrict__ maa_x,
                            const float* __restrict__ maa_vc,
                            const float* __restrict__ q_w1,
                            const float* __restrict__ kv_w1,
                            float* __restrict__ mq_pre,
                            float* __restrict__ t64) {
  __shared__ float sxxx[Hc];
  __shared__ float sxx2[Hc];
  const size_t token = blockIdx.x;
  const int s = (int)(token % Sc), b = (int)(token / Sc);
  const float* xr = x + token * (size_t)Hc;
  const float* xp = (s == 0) ? (shift_in + (size_t)b * Hc) : (x + (token - 1) * (size_t)Hc);
  const float* nr = xo_n + token * (size_t)Hc;
  for (int i = threadIdx.x; i < Hc; i += 256) {
    float xv = xr[i];
    sxxx[i] = xv + (xp[i] - xv) * maa_x[i];
    float nv = nr[i];
    float pv = (s == 0) ? 0.f : xo_n[(token - 1) * (size_t)Hc + i];
    sxx2[i] = nv + (pv - nv) * maa_vc[i];
  }
  __syncthreads();
  const int wv = threadIdx.x >> 5, lane = threadIdx.x & 31;
  for (int o = wv; o < DMIX + 2 * DMIX; o += 8) {          // 96 dot products
    const float* src; const float* wcol; int stride;
    if (o < DMIX) { src = sxxx; wcol = q_w1 + o;           stride = DMIX;     }
    else          { src = sxx2; wcol = kv_w1 + (o - DMIX); stride = 2 * DMIX; }
    float acc = 0.f;
    for (int i = lane; i < Hc; i += 32) acc += src[i] * wcol[(size_t)i * stride];
#pragma unroll
    for (int off = 16; off > 0; off >>= 1) acc += __shfl_xor(acc, off, 32);
    if (lane == 0) {
      float r = tanhf(acc);
      if (o < DMIX) mq_pre[token * DMIX + o] = r;
      else          t64[token * (2 * DMIX) + (o - DMIX)] = r;
    }
  }
}

// ---------------------------------------------------------------------------
// K3: mq/mk/mv = small projections; xq (bf16), k1, v1
// ---------------------------------------------------------------------------
__global__ void mix_kernel(const float* __restrict__ x,
                           const float* __restrict__ shift_in,
                           const float* __restrict__ k_cache,
                           const float* __restrict__ xo_n,
                           const float* __restrict__ maa_q,
                           const float* __restrict__ maa_k,
                           const float* __restrict__ maa_v,
                           const float* __restrict__ q_w2,
                           const float* __restrict__ kv_w2,
                           const float* __restrict__ mq_pre,
                           const float* __restrict__ t64,
                           unsigned short* __restrict__ xq_bf,
                           float* __restrict__ k1,
                           float* __restrict__ v1) {
  __shared__ float smq[DMIX], smk[DMIX], smv[DMIX];
  const size_t token = blockIdx.x;
  const int s = (int)(token % Sc), b = (int)(token / Sc);
  const int tid = threadIdx.x;
  if (tid < DMIX)                 smq[tid]            = mq_pre[token * DMIX + tid];
  else if (tid < 2 * DMIX)        smk[tid - DMIX]     = t64[token * (2 * DMIX) + (tid - DMIX)];
  else if (tid < 3 * DMIX)        smv[tid - 2 * DMIX] = t64[token * (2 * DMIX) + DMIX + (tid - 2 * DMIX)];
  __syncthreads();
  for (int h = tid; h < Hc; h += 256) {
    float mq = 0.f, mk = 0.f, mv = 0.f;
#pragma unroll
    for (int d = 0; d < DMIX; ++d) {
      mq += smq[d] * q_w2[(size_t)d * Hc + h];
      mk += smk[d] * kv_w2[(size_t)d * Hc + h];
      mv += smv[d] * kv_w2[(size_t)(DMIX + d) * Hc + h];
    }
    const size_t idx = token * (size_t)Hc + h;
    float xv = x[idx];
    float xp = (s == 0) ? shift_in[(size_t)b * Hc + h] : x[idx - Hc];
    xq_bf[idx] = f32_to_bf16(xv + (xp - xv) * (maa_q[h] + mq));
    float kc = k_cache[idx];
    float kp = (s == 0) ? 0.f : k_cache[idx - Hc];
    k1[idx] = kc + (kp - kc) * (maa_k[h] + mk);
    float vn = xo_n[idx];
    float vp = (s == 0) ? 0.f : xo_n[idx - Hc];
    v1[idx] = vn + (vp - vn) * (maa_v[h] + mv);
  }
}

// ---------------------------------------------------------------------------
// K4: k = LN(k1 + tanh(k1@key_w1)@key_w2), v likewise -> bf16
// ---------------------------------------------------------------------------
__global__ void mlp2_ln_kernel(const float* __restrict__ k1,
                               const float* __restrict__ v1,
                               const float* __restrict__ key_w1,
                               const float* __restrict__ key_w2,
                               const float* __restrict__ value_w1,
                               const float* __restrict__ value_w2,
                               const float* __restrict__ ln_k_g, const float* __restrict__ ln_k_b,
                               const float* __restrict__ ln_v_g, const float* __restrict__ ln_v_b,
                               unsigned short* __restrict__ k_bf,
                               unsigned short* __restrict__ v_bf) {
  __shared__ float sk[Hc];
  __shared__ float sv[Hc];
  __shared__ float hk[DVAL];
  __shared__ float hv[DVAL];
  __shared__ float red[8];
  const size_t token = blockIdx.x;
  for (int i = threadIdx.x; i < Hc; i += 256) {
    sk[i] = k1[token * (size_t)Hc + i];
    sv[i] = v1[token * (size_t)Hc + i];
  }
  __syncthreads();
  const int wv = threadIdx.x >> 5, lane = threadIdx.x & 31;
  for (int o = wv; o < 2 * DVAL; o += 8) {                 // 256 dots of length H
    const bool isk = o < DVAL;
    const int c = isk ? o : o - DVAL;
    const float* w = (isk ? key_w1 : value_w1) + c;
    const float* src = isk ? sk : sv;
    float acc = 0.f;
    for (int i = lane; i < Hc; i += 32) acc += src[i] * w[(size_t)i * DVAL];
#pragma unroll
    for (int off = 16; off > 0; off >>= 1) acc += __shfl_xor(acc, off, 32);
    if (lane == 0) { if (isk) hk[c] = tanhf(acc); else hv[c] = tanhf(acc); }
  }
  __syncthreads();
  for (int h = threadIdx.x; h < Hc; h += 256) {
    float ak = 0.f, av = 0.f;
#pragma unroll 8
    for (int d = 0; d < DVAL; ++d) {
      ak += hk[d] * key_w2[(size_t)d * Hc + h];
      av += hv[d] * value_w2[(size_t)d * Hc + h];
    }
    sk[h] += ak; sv[h] += av;
  }
  __syncthreads();
  float s1 = 0.f;
  for (int i = threadIdx.x; i < Hc; i += 256) s1 += sk[i];
  float mk = block_sum(s1, red) / (float)Hc;
  float s2 = 0.f;
  for (int i = threadIdx.x; i < Hc; i += 256) { float d = sk[i] - mk; s2 += d * d; }
  float ik = rsqrtf(block_sum(s2, red) / (float)Hc + 1e-5f);
  for (int i = threadIdx.x; i < Hc; i += 256)
    k_bf[token * (size_t)Hc + i] = f32_to_bf16((sk[i] - mk) * ik * ln_k_g[i] + ln_k_b[i]);
  float s3 = 0.f;
  for (int i = threadIdx.x; i < Hc; i += 256) s3 += sv[i];
  float mv = block_sum(s3, red) / (float)Hc;
  float s4 = 0.f;
  for (int i = threadIdx.x; i < Hc; i += 256) { float d = sv[i] - mv; s4 += d * d; }
  float iv = rsqrtf(block_sum(s4, red) / (float)Hc + 1e-5f);
  for (int i = threadIdx.x; i < Hc; i += 256)
    v_bf[token * (size_t)Hc + i] = f32_to_bf16((sv[i] - mv) * iv * ln_v_g[i] + ln_v_b[i]);
}

// ---------------------------------------------------------------------------
// f32 -> bf16 elementwise (weights)
// ---------------------------------------------------------------------------
__global__ void cvt_bf16_kernel(const float* __restrict__ in, unsigned short* __restrict__ out, size_t n) {
  for (size_t i = blockIdx.x * 256ull + threadIdx.x; i < n; i += (size_t)gridDim.x * 256ull)
    out[i] = f32_to_bf16(in[i]);
}

// ---------------------------------------------------------------------------
// WMMA bf16 GEMM:  C[M,N] = A[M,K] @ W[N,K]^T   (both row-major, "NT" layout)
// wave computes a 32x64 slab (8 accumulators) so each B fragment feeds 2 WMMA.
// block = 8 waves = 32 x 512.
// ---------------------------------------------------------------------------
__global__ void gemm_bf16_nt(const unsigned short* __restrict__ A,
                             const unsigned short* __restrict__ W,
                             float* __restrict__ C, int M, int N, int K) {
  const int lane = threadIdx.x & 31;
  const int wave = threadIdx.x >> 5;
  const int mt = blockIdx.x * 32;
  const int nt = blockIdx.y * 512 + wave * 64;
  const int arow = lane & 15;          // A: lane holds matrix row; B: lane holds column
  const int kh   = lane >> 4;          // K half select

  v8f acc[8] = {};
  for (int k0 = 0; k0 < K; k0 += 32) {
    FragAB a0, a1;
    const unsigned short* ap0 = A + (size_t)(mt + arow) * K + k0 + kh * 8;
    const unsigned short* ap1 = ap0 + (size_t)16 * K;
    a0.f4[0] = *(const float4*)(ap0);
    a0.f4[1] = *(const float4*)(ap0 + 16);
    a1.f4[0] = *(const float4*)(ap1);
    a1.f4[1] = *(const float4*)(ap1 + 16);
    __builtin_prefetch(ap0 + 64, 0, 0);              // global_prefetch next K slab
    __builtin_prefetch(ap1 + 64, 0, 0);
#pragma unroll
    for (int s = 0; s < 4; ++s) {
      FragAB bfr;
      const unsigned short* wp = W + (size_t)(nt + s * 16 + arow) * K + k0 + kh * 8;
      bfr.f4[0] = *(const float4*)(wp);
      bfr.f4[1] = *(const float4*)(wp + 16);
      acc[s] = __builtin_amdgcn_wmma_f32_16x16x32_bf16(
          false, a0.v, false, bfr.v, (short)0, acc[s], false, false);
      acc[4 + s] = __builtin_amdgcn_wmma_f32_16x16x32_bf16(
          false, a1.v, false, bfr.v, (short)0, acc[4 + s], false, false);
    }
  }
  const int n = lane & 15;
  const int rbase = (lane >> 4) * 8;
#pragma unroll
  for (int half = 0; half < 2; ++half)
#pragma unroll
    for (int s = 0; s < 4; ++s)
#pragma unroll
      for (int r = 0; r < 8; ++r)
        C[(size_t)(mt + half * 16 + rbase + r) * N + nt + s * 16 + n] = acc[half * 4 + s][r];
}

// ---------------------------------------------------------------------------
// Row LayerNorm -> bf16
// ---------------------------------------------------------------------------
__global__ void ln_rows_to_bf16(const float* __restrict__ in,
                                const float* __restrict__ g, const float* __restrict__ bb,
                                unsigned short* __restrict__ out) {
  __shared__ float red[8];
  const size_t token = blockIdx.x;
  const float* row = in + token * (size_t)Hc;
  float s1 = 0.f;
  for (int i = threadIdx.x; i < Hc; i += 256) s1 += row[i];
  float mean = block_sum(s1, red) / (float)Hc;
  float s2 = 0.f;
  for (int i = threadIdx.x; i < Hc; i += 256) { float d = row[i] - mean; s2 += d * d; }
  float inv = rsqrtf(block_sum(s2, red) / (float)Hc + 1e-5f);
  for (int i = threadIdx.x; i < Hc; i += 256)
    out[token * (size_t)Hc + i] = f32_to_bf16((row[i] - mean) * inv * g[i] + bb[i]);
}

// ---------------------------------------------------------------------------
// Flash attention, WMMA bf16. One wave per (b, head, 16-row query tile).
// Processes 32 keys per step: scores = two 16x16 C tiles, P = dense 16x32
// A fragment, V = 32x64 LDS tile staged with GLOBAL_LOAD_ASYNC_TO_LDS_B128.
// ---------------------------------------------------------------------------
__global__ void flash_attn_kernel(const unsigned short* __restrict__ q,
                                  const unsigned short* __restrict__ k,
                                  const unsigned short* __restrict__ v,
                                  float* __restrict__ o) {
  __shared__ unsigned short lds_p[8][16 * 32];   // P tile (16x32) per wave
  __shared__ unsigned short lds_v[8][32 * KDc];  // V tile (32x64) per wave
  const int w = threadIdx.x >> 5;
  const int lane = threadIdx.x & 31;
  const int qtiles = Sc / 16;
  const int wid = blockIdx.x * 8 + w;
  const int qt = wid % qtiles;
  const int head = (wid / qtiles) % NHc;
  const int b = wid / (qtiles * NHc);
  unsigned short* P  = lds_p[w];
  unsigned short* VT = lds_v[w];
  const float scale = rsqrtf((float)KDc);

  const int col = lane & 15;
  const int kh = lane >> 4;
  const int rbase = kh * 8;

  // Q fragments: 16 rows x 64 dims = two K=32 fragments
  FragAB qa[2];
  {
    const unsigned short* qp = q + ((size_t)b * Sc + qt * 16 + col) * Hc + head * KDc;
#pragma unroll
    for (int t = 0; t < 2; ++t) {
      const unsigned short* p0 = qp + t * 32 + kh * 8;
      qa[t].f4[0] = *(const float4*)(p0);
      qa[t].f4[1] = *(const float4*)(p0 + 16);
    }
  }
  float m_run[8], l_run[8];
#pragma unroll
  for (int r = 0; r < 8; ++r) { m_run[r] = -1e30f; l_run[r] = 0.f; }
  v8f oacc[4] = {};

  const int kvblocks = (qt >> 1) + 1;            // 32-key blocks (causal)
  for (int j32 = 0; j32 < kvblocks; ++j32) {
    // ---- async-stage V block (32 rows x 64 cols bf16): one row per lane
    {
      const unsigned long long gsrc =
          (unsigned long long)(v + ((size_t)b * Sc + j32 * 32 + lane) * Hc + head * KDc);
      const unsigned ldst = (unsigned)(uintptr_t)(VT + lane * KDc);
#pragma unroll
      for (int c2 = 0; c2 < 8; ++c2) {
        asm volatile("global_load_async_to_lds_b128 %0, %1, off"
                     :: "v"(ldst + c2 * 16), "v"(gsrc + c2 * 16) : "memory");
      }
    }
    // ---- scores = Q @ K_block^T : two 16x16 tiles
    v8f sa0 = {}, sa1 = {};
#pragma unroll
    for (int t = 0; t < 2; ++t) {
      FragAB kb0, kb1;
      const unsigned short* kp0 =
          k + ((size_t)b * Sc + j32 * 32 + col) * Hc + head * KDc + t * 32 + kh * 8;
      const unsigned short* kp1 = kp0 + (size_t)16 * Hc;
      kb0.f4[0] = *(const float4*)(kp0);
      kb0.f4[1] = *(const float4*)(kp0 + 16);
      kb1.f4[0] = *(const float4*)(kp1);
      kb1.f4[1] = *(const float4*)(kp1 + 16);
      sa0 = __builtin_amdgcn_wmma_f32_16x16x32_bf16(
          false, qa[t].v, false, kb0.v, (short)0, sa0, false, false);
      sa1 = __builtin_amdgcn_wmma_f32_16x16x32_bf16(
          false, qa[t].v, false, kb1.v, (short)0, sa1, false, false);
    }
    // ---- causal mask + online softmax over 32 columns
    const int gcol0 = j32 * 32 + col;
    float corr[8], e0[8], e1[8];
#pragma unroll
    for (int r = 0; r < 8; ++r) {
      const int grow = qt * 16 + rbase + r;
      float sv0 = sa0[r] * scale;
      float sv1 = sa1[r] * scale;
      if (gcol0 > grow)      sv0 = -1e30f;
      if (gcol0 + 16 > grow) sv1 = -1e30f;
      float mx = fmaxf(sv0, sv1);
#pragma unroll
      for (int off = 1; off < 16; off <<= 1) mx = fmaxf(mx, __shfl_xor(mx, off, 16));
      float mnew = fmaxf(m_run[r], mx);
      float p0 = __expf(sv0 - mnew);
      float p1 = __expf(sv1 - mnew);
      float ps = p0 + p1;
#pragma unroll
      for (int off = 1; off < 16; off <<= 1) ps += __shfl_xor(ps, off, 16);
      float c = __expf(m_run[r] - mnew);
      l_run[r] = l_run[r] * c + ps;
      m_run[r] = mnew;
      e0[r] = p0; e1[r] = p1; corr[r] = c;
    }
#pragma unroll
    for (int s2 = 0; s2 < 4; ++s2)
#pragma unroll
      for (int r = 0; r < 8; ++r) oacc[s2][r] *= corr[r];
    // ---- transpose P (16x32) through LDS
#pragma unroll
    for (int r = 0; r < 8; ++r) {
      P[(rbase + r) * 32 + col]      = f32_to_bf16(e0[r]);
      P[(rbase + r) * 32 + 16 + col] = f32_to_bf16(e1[r]);
    }
    asm volatile("s_wait_dscnt 0"    ::: "memory");  // P tile RAW (same wave)
    asm volatile("s_wait_asynccnt 0" ::: "memory");  // V tile async arrival
    // ---- O += P(16x32) @ V(32x64)   (dense K)
    FragAB pa;
    {
      const unsigned short* pp = P + (lane & 15) * 32 + kh * 8;
      pa.f4[0] = *(const float4*)(pp);
      pa.f4[1] = *(const float4*)(pp + 16);
    }
#pragma unroll
    for (int s2 = 0; s2 < 4; ++s2) {
      FragAB vb;
#pragma unroll
      for (int j2 = 0; j2 < 16; ++j2) vb.u[j2] = VT[(kh * 8 + j2 + (j2 & 8)) * KDc + s2 * 16 + col];
      oacc[s2] = __builtin_amdgcn_wmma_f32_16x16x32_bf16(
          false, pa.v, false, vb.v, (short)0, oacc[s2], false, false);
    }
  }
  // ---- epilogue: o[b, s, head*Kd + d] = oacc / l
#pragma unroll
  for (int s2 = 0; s2 < 4; ++s2)
#pragma unroll
    for (int r = 0; r < 8; ++r)
      o[((size_t)b * Sc + qt * 16 + rbase + r) * Hc + head * KDc + s2 * 16 + col] =
          oacc[s2][r] / l_run[r];
}

// ---------------------------------------------------------------------------
// Launch
// ---------------------------------------------------------------------------
extern "C" void kernel_launch(void* const* d_in, const int* in_sizes, int n_in,
                              void* d_out, int out_size, void* d_ws, size_t ws_size,
                              hipStream_t stream) {
  const float* x        = (const float*)d_in[0];
  const float* shift_in = (const float*)d_in[1];
  const float* xo       = (const float*)d_in[2];
  const float* k_cache  = (const float*)d_in[3];
  const float* maa_x    = (const float*)d_in[4];
  const float* maa_q    = (const float*)d_in[5];
  const float* maa_vc   = (const float*)d_in[6];
  const float* maa_k    = (const float*)d_in[7];
  const float* maa_v    = (const float*)d_in[8];
  const float* q_w1     = (const float*)d_in[9];
  const float* q_w2     = (const float*)d_in[10];
  const float* kv_w1    = (const float*)d_in[11];
  const float* kv_w2    = (const float*)d_in[12];
  const float* key_w1   = (const float*)d_in[13];
  const float* key_w2   = (const float*)d_in[14];
  const float* value_w1 = (const float*)d_in[15];
  const float* value_w2 = (const float*)d_in[16];
  const float* Wq       = (const float*)d_in[17];
  const float* Wo       = (const float*)d_in[18];
  const float* ln_q_g   = (const float*)d_in[19];
  const float* ln_q_b   = (const float*)d_in[20];
  const float* ln_k_g   = (const float*)d_in[21];
  const float* ln_k_b   = (const float*)d_in[22];
  const float* ln_v_g   = (const float*)d_in[23];
  const float* ln_v_b   = (const float*)d_in[24];
  const float* ln_x_g   = (const float*)d_in[25];
  const float* ln_x_b   = (const float*)d_in[26];

  const size_t T  = (size_t)Bc * Sc;      // 4096 tokens
  const size_t TH = T * Hc;               // 8M elements

  char* ws = (char*)d_ws;
  size_t off = 0;
  auto alloc = [&](size_t bytes) -> char* {
    char* p = ws + off;
    off += (bytes + 255) & ~(size_t)255;
    return p;
  };
  float*          xo_n   = (float*)alloc(TH * 4);
  float*          k1     = (float*)alloc(TH * 4);
  float*          v1     = (float*)alloc(TH * 4);
  float*          q_raw  = (float*)alloc(TH * 4);
  float*          o_f32  = (float*)alloc(TH * 4);
  unsigned short* xq_bf  = (unsigned short*)alloc(TH * 2);
  unsigned short* q_bf   = (unsigned short*)alloc(TH * 2);
  unsigned short* k_bf   = (unsigned short*)alloc(TH * 2);
  unsigned short* v_bf   = (unsigned short*)alloc(TH * 2);
  unsigned short* o_bf   = (unsigned short*)alloc(TH * 2);
  unsigned short* Wq_bf  = (unsigned short*)alloc((size_t)Hc * Hc * 2);
  unsigned short* Wo_bf  = (unsigned short*)alloc((size_t)Hc * Hc * 2);
  float*          mq_pre = (float*)alloc(T * DMIX * 4);
  float*          t64    = (float*)alloc(T * 2 * DMIX * 4);

  float* out_o     = (float*)d_out;          // (B,S,H)
  float* out_shift = out_o + TH;             // (B,H)

  rms_shift_kernel<<<(unsigned)T, 256, 0, stream>>>(x, xo, xo_n, out_shift);
  mlp1_kernel<<<(unsigned)T, 256, 0, stream>>>(x, shift_in, xo_n, maa_x, maa_vc,
                                               q_w1, kv_w1, mq_pre, t64);
  mix_kernel<<<(unsigned)T, 256, 0, stream>>>(x, shift_in, k_cache, xo_n,
                                              maa_q, maa_k, maa_v, q_w2, kv_w2,
                                              mq_pre, t64, xq_bf, k1, v1);
  mlp2_ln_kernel<<<(unsigned)T, 256, 0, stream>>>(k1, v1, key_w1, key_w2,
                                                  value_w1, value_w2,
                                                  ln_k_g, ln_k_b, ln_v_g, ln_v_b,
                                                  k_bf, v_bf);
  cvt_bf16_kernel<<<2048, 256, 0, stream>>>(Wq, Wq_bf, (size_t)Hc * Hc);
  cvt_bf16_kernel<<<2048, 256, 0, stream>>>(Wo, Wo_bf, (size_t)Hc * Hc);

  dim3 ggrid((unsigned)(T / 32), Hc / 512);
  gemm_bf16_nt<<<ggrid, 256, 0, stream>>>(xq_bf, Wq_bf, q_raw, (int)T, Hc, Hc);
  ln_rows_to_bf16<<<(unsigned)T, 256, 0, stream>>>(q_raw, ln_q_g, ln_q_b, q_bf);

  const unsigned attn_blocks = (Bc * NHc * (Sc / 16)) / 8;   // 1024
  flash_attn_kernel<<<attn_blocks, 256, 0, stream>>>(q_bf, k_bf, v_bf, o_f32);

  ln_rows_to_bf16<<<(unsigned)T, 256, 0, stream>>>(o_f32, ln_x_g, ln_x_b, o_bf);
  gemm_bf16_nt<<<ggrid, 256, 0, stream>>>(o_bf, Wo_bf, out_o, (int)T, Hc, Hc);
}
